// ContrastiveRobertaMoCoKnnBert_29764123361921
// MI455X (gfx1250) — matrix operand order repaired and checked by reading, not compiled, not run on hardware
//
#include <hip/hip_runtime.h>
#include <math.h>

// Sizes (fixed by the reference)
#define BATCH 128
#define QK 32768
#define HID 768
#define TOPK 25
#define ENDK 25
#define TEMP 0.07f

typedef __attribute__((ext_vector_type(16))) _Float16 v16h;
typedef __attribute__((ext_vector_type(8)))  _Float16 v8h;
typedef __attribute__((ext_vector_type(8)))  float    v8f;

// ---------------------------------------------------------------------------
// helpers
// ---------------------------------------------------------------------------
__device__ __forceinline__ unsigned mapf(float f) {
    unsigned u = __float_as_uint(f);
    return (u & 0x80000000u) ? ~u : (u | 0x80000000u);  // monotone float->uint
}
__device__ __forceinline__ float invmapf(unsigned k) {
    unsigned u = (k & 0x80000000u) ? (k & 0x7FFFFFFFu) : ~k;
    return __uint_as_float(u);
}

__device__ __forceinline__ unsigned blockMaxU(unsigned v, unsigned* s) {
    int t = threadIdx.x;
    s[t] = v; __syncthreads();
    for (int o = 128; o > 0; o >>= 1) {
        if (t < o) s[t] = (s[t] > s[t + o]) ? s[t] : s[t + o];
        __syncthreads();
    }
    unsigned r = s[0]; __syncthreads(); return r;
}
__device__ __forceinline__ int blockSumI(int v, int* s) {
    int t = threadIdx.x;
    s[t] = v; __syncthreads();
    for (int o = 128; o > 0; o >>= 1) {
        if (t < o) s[t] += s[t + o];
        __syncthreads();
    }
    int r = s[0]; __syncthreads(); return r;
}
__device__ __forceinline__ float blockSumF(float v, float* s) {
    int t = threadIdx.x;
    s[t] = v; __syncthreads();
    for (int o = 128; o > 0; o >>= 1) {
        if (t < o) s[t] += s[t + o];
        __syncthreads();
    }
    float r = s[0]; __syncthreads(); return r;
}

// ---------------------------------------------------------------------------
// f32 -> f16 bulk convert (8 elements / thread, vectorized)
// ---------------------------------------------------------------------------
__global__ void cvt_f16_kernel(const float* __restrict__ src, _Float16* __restrict__ dst,
                               long n) {
    long i = ((long)blockIdx.x * blockDim.x + threadIdx.x) * 8;
    if (i + 8 <= n) {
        float4 f0 = *reinterpret_cast<const float4*>(src + i);
        float4 f1 = *reinterpret_cast<const float4*>(src + i + 4);
        v8h h;
        h[0] = (_Float16)f0.x; h[1] = (_Float16)f0.y;
        h[2] = (_Float16)f0.z; h[3] = (_Float16)f0.w;
        h[4] = (_Float16)f1.x; h[5] = (_Float16)f1.y;
        h[6] = (_Float16)f1.z; h[7] = (_Float16)f1.w;
        *reinterpret_cast<v8h*>(dst + i) = h;
    }
}

// ---------------------------------------------------------------------------
// Head GEMM:  Out[128,N] = op(A[128,768] @ W[768,N] + bias), op = tanh or id
// One wave per 16x16 tile; 8 waves/block cover the 128 rows; grid.x = N/16.
// ---------------------------------------------------------------------------
__global__ void gemm_xw_wmma(const float* __restrict__ A, const float* __restrict__ W,
                             const float* __restrict__ bias, float* __restrict__ Out,
                             int N, int applyTanh) {
    const int wave = threadIdx.x >> 5;      // 0..7 -> m tile
    const int lane = threadIdx.x & 31;
    const int half = lane >> 4;
    const int lrow = lane & 15;
    const int m0 = wave * 16;
    const int n0 = blockIdx.x * 16;
    const float* arow = A + (m0 + lrow) * HID;
    const int col = n0 + lrow;

    v8f acc = {0.f, 0.f, 0.f, 0.f, 0.f, 0.f, 0.f, 0.f};
    for (int kk = 0; kk < HID; kk += 32) {
        v16h a, bf;
        const int ka = kk + half * 16;      // A layout: lanes0-15 K0-15, lanes16-31 K16-31
        #pragma unroll
        for (int j = 0; j < 8; j++) {
            a[j]     = (_Float16)arow[ka + j];
            a[8 + j] = (_Float16)arow[ka + 8 + j];
        }
        const int kb = kk + half * 16;      // B layout (same contiguous order as A): lanes0-15 K0-15, lanes16-31 K16-31
        #pragma unroll
        for (int j = 0; j < 16; j++) bf[j] = (_Float16)W[(kb + j) * N + col];
        acc = __builtin_amdgcn_wmma_f32_16x16x32_f16(false, a, false, bf,
                                                     (short)0, acc, false, false);
    }
    #pragma unroll
    for (int r = 0; r < 8; r++) {
        const int orow = m0 + r + half * 8; // ISA C layout
        float v = acc[r] + bias[col];
        if (applyTanh) v = tanhf(v);
        Out[orow * N + col] = v;
    }
}

// ---------------------------------------------------------------------------
// Big GEMM:  cos[128,32768] = Q16[128,768] @ FQ16[32768,768]^T (f16 in, f32 acc)
// Both operands pre-converted to f16, so the K-loop is pure b128 loads + WMMA:
// A fragment = two 16B v8h loads concatenated; B fragment = one 32B v16h load.
// ---------------------------------------------------------------------------
__global__ void gemm_qfq_wmma(const _Float16* __restrict__ Q, const _Float16* __restrict__ FQ,
                              float* __restrict__ C) {
    const int wave = threadIdx.x >> 5;
    const int lane = threadIdx.x & 31;
    const int half = lane >> 4;
    const int lrow = lane & 15;
    const int m0 = wave * 16;
    const int n0 = blockIdx.x * 16;
    const _Float16* arow = Q + (size_t)(m0 + lrow) * HID;
    const _Float16* brow = FQ + (size_t)(n0 + lrow) * HID;

    v8f acc = {0.f, 0.f, 0.f, 0.f, 0.f, 0.f, 0.f, 0.f};
    for (int kk = 0; kk < HID; kk += 32) {
        const int ka = kk + half * 16;
        v8h alo = *reinterpret_cast<const v8h*>(arow + ka);
        v8h ahi = *reinterpret_cast<const v8h*>(arow + ka + 8);
        v16h a = __builtin_shufflevector(alo, ahi,
                                         0, 1, 2, 3, 4, 5, 6, 7,
                                         8, 9, 10, 11, 12, 13, 14, 15);
        const int kb = kk + half * 16;
        v16h bf = *reinterpret_cast<const v16h*>(brow + kb);
        acc = __builtin_amdgcn_wmma_f32_16x16x32_f16(false, a, false, bf,
                                                     (short)0, acc, false, false);
    }
    #pragma unroll
    for (int r = 0; r < 8; r++) {
        const int orow = m0 + r + half * 8;
        C[(size_t)orow * QK + n0 + lrow] = acc[r];
    }
}

// ---------------------------------------------------------------------------
// cls output head: logits[128,2] = H[128,768] @ Wout[768,2] + b
// ---------------------------------------------------------------------------
__global__ void cls_out_kernel(const float* __restrict__ H, const float* __restrict__ W,
                               const float* __restrict__ bias, float* __restrict__ logits) {
    const int b = blockIdx.x, t = threadIdx.x;
    __shared__ float s0[256], s1[256];
    float a0 = 0.f, a1 = 0.f;
    for (int k = t; k < HID; k += 256) {
        float h = H[b * HID + k];
        a0 += h * W[k * 2 + 0];
        a1 += h * W[k * 2 + 1];
    }
    s0[t] = a0; s1[t] = a1; __syncthreads();
    for (int o = 128; o > 0; o >>= 1) {
        if (t < o) { s0[t] += s0[t + o]; s1[t] += s1[t + o]; }
        __syncthreads();
    }
    if (t == 0) {
        logits[2 * b + 0] = s0[0] + bias[0];
        logits[2 * b + 1] = s1[0] + bias[1];
    }
}

// row L2 normalize; emit f16 copy for the WMMA GEMM
__global__ void rownorm_cvt_kernel(const float* __restrict__ Y, _Float16* __restrict__ Q16) {
    const int b = blockIdx.x, t = threadIdx.x;
    __shared__ float sf[256];
    float a = 0.f;
    for (int k = t; k < HID; k += 256) { float v = Y[b * HID + k]; a += v * v; }
    sf[t] = a; __syncthreads();
    for (int o = 128; o > 0; o >>= 1) { if (t < o) sf[t] += sf[t + o]; __syncthreads(); }
    __shared__ float nrm;
    if (t == 0) nrm = sqrtf(sf[0]);
    __syncthreads();
    for (int k = t; k < HID; k += 256)
        Q16[b * HID + k] = (_Float16)(Y[b * HID + k] / nrm);
}

// pos_min / neg_min (NUM_LABELS==2 -> one queue count suffices)
__global__ void count_min_kernel(const int* __restrict__ labels, const int* __restrict__ lq,
                                 int* __restrict__ out) {
    const int t = threadIdx.x;
    __shared__ int si[256];
    int c = 0;
    for (int i = t; i < QK; i += 256) c += (lq[i] == 0);
    int c0 = blockSumI(c, si);
    int pmin = 0x7FFFFFFF, nmin = 0x7FFFFFFF;
    for (int b = t; b < BATCH; b += 256) {
        int pc = (labels[b] == 0) ? c0 : (QK - c0);
        pmin = min(pmin, pc);
        nmin = min(nmin, QK - pc);
    }
    si[t] = pmin; __syncthreads();
    for (int o = 128; o > 0; o >>= 1) { if (t < o) si[t] = min(si[t], si[t + o]); __syncthreads(); }
    if (t == 0) out[0] = si[0];
    __syncthreads();
    si[t] = nmin; __syncthreads();
    for (int o = 128; o > 0; o >>= 1) { if (t < o) si[t] = min(si[t], si[t + o]); __syncthreads(); }
    if (t == 0) out[1] = si[0];
}

// MSB-first radix select: value key at descending 0-indexed `rank` within class
__device__ unsigned radixSelectDesc(const float* __restrict__ row, const int* __restrict__ lq,
                                    int lab, int wantPos, int rank,
                                    unsigned* hist, unsigned* pPrefix, int* pRem) {
    const int t = threadIdx.x;
    if (t == 0) { *pPrefix = 0u; *pRem = rank + 1; }
    __syncthreads();
    for (int shift = 24; shift >= 0; shift -= 8) {
        hist[t] = 0u; __syncthreads();
        const unsigned pref = *pPrefix;
        const unsigned pmask = (shift == 24) ? 0u : (0xFFFFFFFFu << (shift + 8));
        for (int k = t; k < QK; k += 256) {
            if ((lq[k] == lab) == (wantPos != 0)) {
                unsigned key = mapf(row[k]);
                if ((key & pmask) == pref) atomicAdd(&hist[(key >> shift) & 255u], 1u);
            }
        }
        __syncthreads();
        if (t == 0) {
            int rem = *pRem, d = 255;
            for (; d > 0; d--) { int c = (int)hist[d]; if (rem <= c) break; rem -= c; }
            *pRem = rem;
            *pPrefix = pref | ((unsigned)d << shift);
        }
        __syncthreads();
    }
    unsigned r = *pPrefix; __syncthreads(); return r;
}

// per-row: top-25 positives, end-25 positives (ranks pos_min-25..pos_min-1),
// S_b = sum over top neg_min negatives of exp(v/T). Tie-exact.
__global__ void select_kernel(const float* __restrict__ cosm, const int* __restrict__ labels,
                              const int* __restrict__ lq, const int* __restrict__ minv,
                              float* __restrict__ topv, float* __restrict__ endv,
                              float* __restrict__ Sb) {
    const int b = blockIdx.x, t = threadIdx.x;
    const float* row = cosm + (size_t)b * QK;
    const int lab = labels[b];
    const int pos_min = minv[0], neg_min = minv[1];

    __shared__ unsigned su[256];
    __shared__ int si[256];
    __shared__ float sf[256];
    __shared__ unsigned hist[256];
    __shared__ unsigned sPrefix;
    __shared__ int sRem;

    // ---- top-25 positives via iterative max extraction ----
    unsigned cur = 0xFFFFFFFFu; int eqCnt = 0, eqUsed = 0;
    for (int i = 0; i < TOPK; i++) {
        float val;
        if (eqUsed < eqCnt) { val = invmapf(cur); eqUsed++; }
        else {
            unsigned lm = 0u;
            for (int k = t; k < QK; k += 256)
                if (lq[k] == lab) { unsigned key = mapf(row[k]); if (key < cur && key > lm) lm = key; }
            unsigned m = blockMaxU(lm, su);
            int lc = 0;
            for (int k = t; k < QK; k += 256)
                if (lq[k] == lab && mapf(row[k]) == m) lc++;
            int e = blockSumI(lc, si);
            cur = m; eqCnt = e; eqUsed = 1; val = invmapf(m);
        }
        if (t == 0) topv[b * TOPK + i] = val;
    }

    // ---- end-25 positives: radix select rank r0 = pos_min - 25 ----
    int r0 = pos_min - ENDK; if (r0 < 0) r0 = 0;
    unsigned K0 = radixSelectDesc(row, lq, lab, 1, r0, hist, &sPrefix, &sRem);
    int lg = 0, le = 0;
    for (int k = t; k < QK; k += 256)
        if (lq[k] == lab) { unsigned key = mapf(row[k]); lg += (key > K0); le += (key == K0); }
    int g = blockSumI(lg, si);
    int e = blockSumI(le, si);
    cur = K0; eqCnt = e; eqUsed = e;   // ties at K0 are covered by the r < g+e branch
    for (int j = 0; j < ENDK; j++) {
        int r = r0 + j; float val;
        if (r < g + e) val = invmapf(K0);
        else if (eqUsed < eqCnt) { val = invmapf(cur); eqUsed++; }
        else {
            unsigned lm = 0u;
            for (int k = t; k < QK; k += 256)
                if (lq[k] == lab) { unsigned key = mapf(row[k]); if (key < cur && key > lm) lm = key; }
            unsigned m = blockMaxU(lm, su);
            int lc = 0;
            for (int k = t; k < QK; k += 256)
                if (lq[k] == lab && mapf(row[k]) == m) lc++;
            int e2 = blockSumI(lc, si);
            cur = m; eqCnt = e2; eqUsed = 1; val = invmapf(m);
        }
        if (t == 0) endv[b * ENDK + j] = val;
    }

    // ---- negatives: S_b over top neg_min ----
    int rn = neg_min - 1; if (rn < 0) rn = 0;
    unsigned Kn = radixSelectDesc(row, lq, lab, 0, rn, hist, &sPrefix, &sRem);
    int lgn = 0; float ls = 0.f;
    for (int k = t; k < QK; k += 256) {
        if (lq[k] != lab) {
            float v = row[k]; unsigned key = mapf(v);
            if (key > Kn) { lgn++; ls += expf(v / TEMP); }
        }
    }
    int gn = blockSumI(lgn, si);
    float S = blockSumF(ls, sf);
    S += (float)(neg_min - gn) * expf(invmapf(Kn) / TEMP);
    if (t == 0) Sb[b] = S;
}

// final: loss = 0.1 * loss_con + 0.9 * loss_cls
__global__ void final_kernel(const float* __restrict__ logits, const int* __restrict__ labels,
                             const float* __restrict__ topv, const float* __restrict__ endv,
                             const float* __restrict__ Sb, float* __restrict__ out) {
    const int t = threadIdx.x;
    __shared__ float sf[256];
    float lc = 0.f;
    for (int b = t; b < BATCH; b += 256) {
        float l0 = logits[2 * b], l1 = logits[2 * b + 1];
        float m = fmaxf(l0, l1);
        float lse = m + logf(expf(l0 - m) + expf(l1 - m));
        lc += lse - (labels[b] ? l1 : l0);
    }
    float loss_cls = blockSumF(lc, sf) / (float)BATCH;
    float cc = 0.f;
    for (int i = t; i < BATCH * (TOPK + ENDK); i += 256) {
        int b = i / (TOPK + ENDK), j = i % (TOPK + ENDK);
        float p = (j < TOPK) ? topv[b * TOPK + j] : endv[b * ENDK + (j - TOPK)];
        float z = p / TEMP;
        cc += logf(expf(z) + Sb[b]) - z;
    }
    float loss_con = blockSumF(cc, sf) / (float)(BATCH * (TOPK + ENDK));
    if (t == 0) out[0] = 0.1f * loss_con + 0.9f * loss_cls;
}

// ---------------------------------------------------------------------------
extern "C" void kernel_launch(void* const* d_in, const int* in_sizes, int n_in,
                              void* d_out, int out_size, void* d_ws, size_t ws_size,
                              hipStream_t stream) {
    const float* pooled = (const float*)d_in[0];
    const int*   labels = (const int*)d_in[1];
    const float* fq     = (const float*)d_in[2];
    const int*   lq     = (const int*)d_in[3];
    const float* cdw = (const float*)d_in[4];
    const float* cdb = (const float*)d_in[5];
    const float* cow = (const float*)d_in[6];
    const float* cob = (const float*)d_in[7];
    const float* ndw = (const float*)d_in[8];
    const float* ndb = (const float*)d_in[9];
    const float* now_ = (const float*)d_in[10];
    const float* nob  = (const float*)d_in[11];

    // workspace layout (~68 MB total, 64B-aligned sections)
    char* ws = (char*)d_ws;
    _Float16* fq16 = (_Float16*)ws;                       // 32768*768 halves = 50331648 B
    _Float16* q16  = (_Float16*)(ws + 50331648);          // 128*768 halves  = 196608 B
    float* cosm    = (float*)(ws + 50528256);             // 128*32768 floats
    float* h1      = cosm + (size_t)BATCH * QK;           // 128*768
    float* y       = h1 + BATCH * HID;                    // 128*768
    float* hcls    = y + BATCH * HID;                     // 128*768
    float* logits  = hcls + BATCH * HID;                  // 256
    int*   minv    = (int*)(logits + 256);                // 2 (padded)
    float* topv    = (float*)(minv + 16);                 // 128*25
    float* endv    = topv + BATCH * TOPK;                 // 128*25
    float* Sbuf    = endv + BATCH * ENDK;                 // 128

    dim3 blk(256);
    const long nfq = (long)QK * HID;                      // 25165824, divisible by 8
    // queue f32 -> f16 (one memory-bound pass; removes all cvt from the GEMM hot loop)
    cvt_f16_kernel<<<(unsigned)(nfq / 8 / 256), blk, 0, stream>>>(fq, fq16, nfq);
    // cls head
    gemm_xw_wmma<<<HID / 16, blk, 0, stream>>>(pooled, cdw, cdb, hcls, HID, 1);
    cls_out_kernel<<<BATCH, blk, 0, stream>>>(hcls, cow, cob, logits);
    // con head -> liner_q (f16)
    gemm_xw_wmma<<<HID / 16, blk, 0, stream>>>(pooled, ndw, ndb, h1, HID, 1);
    gemm_xw_wmma<<<HID / 16, blk, 0, stream>>>(h1, now_, nob, y, HID, 0);
    rownorm_cvt_kernel<<<BATCH, blk, 0, stream>>>(y, q16);
    // cos_sim = liner_q @ fq^T  (dominant GEMM, pure f16 WMMA loop)
    gemm_qfq_wmma<<<QK / 16, blk, 0, stream>>>(q16, fq16, cosm);
    // selection + loss
    count_min_kernel<<<1, blk, 0, stream>>>(labels, lq, minv);
    select_kernel<<<BATCH, blk, 0, stream>>>(cosm, labels, lq, minv, topv, endv, Sbuf);
    final_kernel<<<1, blk, 0, stream>>>(logits, labels, topv, endv, Sbuf, (float*)d_out);
}
